// VisionTransformer_15908558864433
// MI455X (gfx1250) — matrix operand order repaired and checked by reading, not compiled, not run on hardware
//
#include <hip/hip_runtime.h>
#include <hip/hip_bf16.h>

// ============================================================================
// ViT forward for MI455X (gfx1250, wave32, WMMA, async-to-LDS).
//  - `tok`/vocab argmin is dead code for logits -> skipped entirely.
//  - All big GEMMs: f16 operands, v_wmma_f32_16x16x32_f16, f32 accumulate.
//  - GEMM tile 64x128, ping-pong LDS, GLOBAL_LOAD_ASYNC_TO_LDS_B128 with
//    threshold waits (s_wait_asynccnt 3) so copies of chunk k+1 overlap the
//    4 WMMAs of chunk k.
//  - Logits GEMM (2048x65536x512, 512MB out) uses non-temporal stores so the
//    store stream does not evict the 67MB f16 outp_w from the 192MB L2.
// ============================================================================

typedef __attribute__((ext_vector_type(4)))  _Float16 v4h;
typedef __attribute__((ext_vector_type(8)))  _Float16 v8h;
typedef __attribute__((ext_vector_type(16))) _Float16 v16h;
typedef __attribute__((ext_vector_type(8)))  float    v8f;
typedef __attribute__((ext_vector_type(4)))  int      v4i;

#if __has_builtin(__builtin_amdgcn_global_load_async_to_lds_b128)
#define HAVE_ASYNC 1
#else
#define HAVE_ASYNC 0
#endif

typedef __attribute__((address_space(1))) v4i gas_v4i;
typedef __attribute__((address_space(3))) v4i las_v4i;

__device__ __forceinline__ void async16(_Float16* lds, const _Float16* g) {
#if HAVE_ASYNC
  __builtin_amdgcn_global_load_async_to_lds_b128((gas_v4i*)g, (las_v4i*)lds, 0, 0);
#else
  *(v8h*)lds = *(const v8h*)g;   // fallback: through VGPRs
#endif
}

template <int N>
__device__ __forceinline__ void wait_async_le() {
#if HAVE_ASYNC
#if __has_builtin(__builtin_amdgcn_s_wait_asynccnt)
  __builtin_amdgcn_s_wait_asynccnt(N);
#else
  asm volatile("s_wait_asynccnt %0" :: "i"(N) : "memory");
#endif
#endif
}

__device__ __forceinline__ v16h cat8(v8h lo, v8h hi) {
  return __builtin_shufflevector(lo, hi, 0,1,2,3,4,5,6,7,8,9,10,11,12,13,14,15);
}

__device__ __forceinline__ v8f wmma_f16(v16h a, v16h b, v8f c) {
  return __builtin_amdgcn_wmma_f32_16x16x32_f16(false, a, false, b, (short)0, c, false, false);
}

#define NTOK 2048
#define DM   512
#define SEQ  256

// ---------------------------------------------------------------------------
// f32 -> f16 streaming convert (n multiple of 4)
// ---------------------------------------------------------------------------
__global__ void cvt_k(const float* __restrict__ s, _Float16* __restrict__ d, int n) {
  int i = (blockIdx.x * blockDim.x + threadIdx.x) * 4;
  if (i >= n) return;
  float4 v = *(const float4*)(s + i);
  v4h h = { (_Float16)v.x, (_Float16)v.y, (_Float16)v.z, (_Float16)v.w };
  *(v4h*)(d + i) = h;
}

// ---------------------------------------------------------------------------
// Tokenizer (vocab argmin is dead code for the output)
// ---------------------------------------------------------------------------
__global__ void tokenize_k(const float* __restrict__ patches,
                           const float* __restrict__ masked_token,
                           const float* __restrict__ partial_params,
                           float* __restrict__ proc) {
  int n = blockIdx.x * blockDim.x + threadIdx.x;
  if (n >= NTOK) return;
  const float* p = patches + (size_t)n * 16;
  float v[16];
  bool fully = true, any_mask = false, any_bin = false;
  int first = -1;
  #pragma unroll
  for (int i = 0; i < 16; ++i) {
    v[i] = p[i];
    bool m = fabsf(v[i] - 0.5f) < 0.1f;
    any_mask |= m; bool bin = !m; any_bin |= bin; fully &= m;
    if (bin && first < 0) first = i;
  }
  bool partial = (!fully) && any_mask && any_bin;
  int fp = first < 0 ? 0 : first;
  int chosen = (int)roundf(v[fp]);
  chosen = chosen < 0 ? 0 : (chosen > 1 ? 1 : chosen);
  int ptok = fp * 2 + chosen;
  float* o = proc + (size_t)n * 16;
  #pragma unroll
  for (int i = 0; i < 16; ++i)
    o[i] = fully ? masked_token[i] : (partial ? partial_params[ptok * 16 + i] : v[i]);
}

// ---------------------------------------------------------------------------
// Embed (K=16, trivial)
// ---------------------------------------------------------------------------
__global__ void embed_k(const float* __restrict__ proc, const float* __restrict__ emb_w,
                        const float* __restrict__ emb_b, float* __restrict__ x) {
  int gid = blockIdx.x * blockDim.x + threadIdx.x;   // over 2048*512
  int n = gid >> 9, d = gid & 511;
  const float* pr = proc + (size_t)n * 16;
  const float* wr = emb_w + (size_t)d * 16;
  float s = emb_b[d];
  #pragma unroll
  for (int k = 0; k < 16; ++k) s += pr[k] * wr[k];
  x[(size_t)n * DM + d] = s;
}

// ---------------------------------------------------------------------------
// LayerNorm: one wave32 per row of 512; writes f16 (GEMM A operand)
// ---------------------------------------------------------------------------
__global__ void layernorm_k(const float* __restrict__ x, const float* __restrict__ g,
                            const float* __restrict__ b, _Float16* __restrict__ y) {
  int row  = blockIdx.x * 8 + (threadIdx.x >> 5);
  int lane = threadIdx.x & 31;
  const float* xr = x + (size_t)row * DM;
  float vals[16];
  float s = 0.f;
  #pragma unroll
  for (int i = 0; i < 16; ++i) { vals[i] = xr[lane + 32 * i]; s += vals[i]; }
  #pragma unroll
  for (int o = 16; o >= 1; o >>= 1) s += __shfl_xor(s, o, 32);
  float mean = s * (1.0f / DM);
  float vs = 0.f;
  #pragma unroll
  for (int i = 0; i < 16; ++i) { float d = vals[i] - mean; vs += d * d; }
  #pragma unroll
  for (int o = 16; o >= 1; o >>= 1) vs += __shfl_xor(vs, o, 32);
  float r = rsqrtf(vs * (1.0f / DM) + 1e-5f);
  _Float16* yr = y + (size_t)row * DM;
  #pragma unroll
  for (int i = 0; i < 16; ++i) {
    int c = lane + 32 * i;
    yr[c] = (_Float16)((vals[i] - mean) * r * g[c] + b[c]);
  }
}

// ---------------------------------------------------------------------------
// WMMA GEMM, all-f16:  out = act(A[M,K] @ W[N,K]^T + bias) (+res)
// block 256 (8 waves), tile 64x128 (wave = 16x64 = 4 WMMA tiles), K chunks
// of 32, ping-pong LDS, async-to-LDS with threshold waits:
//   barrier; issue chunk ch+1 (3 ops); s_wait_asynccnt 3 (chunk ch landed);
//   barrier; 4 WMMAs on chunk ch   -> copies overlap compute.
// grid = (M/64, N/128). act 1 = exact GELU. Ch!=null -> f16 out, else f32
// (nt -> non-temporal stores).
// ---------------------------------------------------------------------------
__global__ void gemm_k(const _Float16* __restrict__ A, const _Float16* __restrict__ W,
                       const float* __restrict__ bias, const float* __restrict__ res,
                       float* __restrict__ Cf, _Float16* __restrict__ Ch,
                       int M, int N, int K, int act, int nt) {
  __shared__ __align__(16) _Float16 sA[2][64 * 40];    // 64 rows x 32 (pad 40)
  __shared__ __align__(16) _Float16 sW[2][128 * 40];   // 128 rows x 32 (pad 40)
  const int t = threadIdx.x, w = t >> 5, lane = t & 31;
  const int grp = lane >> 4, l15 = lane & 15;
  const int rg = w >> 1, cg4 = (w & 1) * 4;            // wave: rows rg*16, cols cg4*16..+63
  const int m0 = blockIdx.x * 64, n0 = blockIdx.y * 128;

  // copy slots: A = 1x16B/thread, W = 2x16B/thread per chunk
  const int rA = t >> 2,      cA = (t & 3) * 8;
  const int rW0 = t >> 2,     cW0 = (t & 3) * 8;       // idx = t
  const int rW1 = (t + 256) >> 2, cW1 = (t & 3) * 8;   // idx = t + 256
  const _Float16* gA  = A + (size_t)(m0 + rA) * K + cA;
  const _Float16* gW0 = W + (size_t)(n0 + rW0) * K + cW0;
  const _Float16* gW1 = W + (size_t)(n0 + rW1) * K + cW1;
  const int lofsA  = rA * 40 + cA;
  const int lofsW0 = rW0 * 40 + cW0;
  const int lofsW1 = rW1 * 40 + cW1;

  const int nch = K >> 5;
  async16(&sA[0][lofsA], gA);                          // prologue: chunk 0 -> buf 0
  async16(&sW[0][lofsW0], gW0);
  async16(&sW[0][lofsW1], gW1);

  v8f acc[4] = {};
  for (int ch = 0; ch < nch; ++ch) {
    const int buf = ch & 1;
    __syncthreads();                                   // all waves done reading buf^1
    if (ch + 1 < nch) {                                // overlap: copy chunk ch+1
      int k1 = (ch + 1) << 5;
      async16(&sA[buf ^ 1][lofsA], gA + k1);
      async16(&sW[buf ^ 1][lofsW0], gW0 + k1);
      async16(&sW[buf ^ 1][lofsW1], gW1 + k1);
      wait_async_le<3>();                              // chunk ch retired, ch+1 in flight
    } else {
      wait_async_le<0>();
    }
    __syncthreads();                                   // chunk ch visible to all waves
    const _Float16* ap = &sA[buf][(rg * 16 + l15) * 40 + grp * 8];
    v16h af = cat8(*(const v8h*)ap, *(const v8h*)(ap + 16));
    #pragma unroll
    for (int tt = 0; tt < 4; ++tt) {
      const _Float16* bp = &sW[buf][((cg4 + tt) * 16 + l15) * 40 + grp * 8];
      v16h bf = cat8(*(const v8h*)bp, *(const v8h*)(bp + 16));
      acc[tt] = wmma_f16(af, bf, acc[tt]);
    }
  }

  #pragma unroll
  for (int j = 0; j < 8; ++j) {
    int m = m0 + rg * 16 + j + 8 * grp;
    #pragma unroll
    for (int tt = 0; tt < 4; ++tt) {
      int n = n0 + (cg4 + tt) * 16 + l15;
      float v = acc[tt][j] + bias[n];
      if (act == 1) v = 0.5f * v * (1.f + erff(v * 0.70710678118654752f));
      size_t off = (size_t)m * N + n;
      if (res) v += res[off];
      if (Ch) Ch[off] = (_Float16)v;
      else if (nt) __builtin_nontemporal_store(v, Cf + off);
      else Cf[off] = v;
    }
  }
}

// ---------------------------------------------------------------------------
// Fused attention per (batch, head, 32-query slab), f16 qkv in/out.
// Q/K staged via async-to-LDS; V staged transposed (scalar f16 copies).
// grid = (B*H=64, S/32=8), block 256. Static LDS = 64000 B.
// ---------------------------------------------------------------------------
__global__ void attention_k(const _Float16* __restrict__ qkv,   // [2048,1536] f16
                            const float* __restrict__ rel,      // [256,256] f32
                            _Float16* __restrict__ out) {       // [2048,512] f16
  __shared__ __align__(16) _Float16 sQ[32 * 72];
  __shared__ __align__(16) _Float16 sKV[64 * 72];
  __shared__ float sS[32 * 260];
  __shared__ __align__(16) _Float16 sP[32 * 264];

  const int bh = blockIdx.x;
  const int b = bh >> 3, h = bh & 7;
  const int q0 = blockIdx.y * 32;
  const int t = threadIdx.x, w = t >> 5, lane = t & 31;
  const int grp = lane >> 4, l15 = lane & 15;
  const int rg = w >> 2, wc = w & 3;
  const size_t rowbase = (size_t)b * SEQ;

  // ---- stage Q (32x64, one async b128 per thread) ----
  {
    int r = t >> 3, c8 = (t & 7) * 8;
    async16(&sQ[r * 72 + c8], qkv + (rowbase + q0 + r) * 1536 + h * 64 + c8);
  }

  // ---- scores = Q K^T * 0.125 + rel ----
  for (int kc = 0; kc < 4; ++kc) {
    __syncthreads();                                   // prior sKV reads done
    #pragma unroll
    for (int i = 0; i < 2; ++i) {
      int idx = t + i * 256;
      int r = idx >> 3, c8 = (idx & 7) * 8;            // r=key, c8=dim
      async16(&sKV[r * 72 + c8],
              qkv + (rowbase + kc * 64 + r) * 1536 + DM + h * 64 + c8);
    }
    wait_async_le<0>();
    __syncthreads();
    v8f acc = {};
    #pragma unroll
    for (int sub = 0; sub < 2; ++sub) {
      const _Float16* ap = &sQ[(rg * 16 + l15) * 72 + sub * 32 + grp * 8];
      v16h af = cat8(*(const v8h*)ap, *(const v8h*)(ap + 16));
      const _Float16* bp = &sKV[(wc * 16 + l15) * 72 + sub * 32 + grp * 8];
      v16h bf = cat8(*(const v8h*)bp, *(const v8h*)(bp + 16));
      acc = wmma_f16(af, bf, acc);
    }
    #pragma unroll
    for (int j = 0; j < 8; ++j) {
      int m = rg * 16 + j + 8 * grp;
      int n = kc * 64 + wc * 16 + l15;
      sS[m * 260 + n] = acc[j] * 0.125f + rel[(size_t)(q0 + m) * SEQ + n];
    }
  }
  __syncthreads();

  // ---- softmax, wave w owns rows w*4..w*4+3 ----
  for (int rr = 0; rr < 4; ++rr) {
    int r = w * 4 + rr;
    float* row = &sS[r * 260];
    float mx = -1e30f;
    for (int i = lane; i < SEQ; i += 32) mx = fmaxf(mx, row[i]);
    #pragma unroll
    for (int o = 16; o >= 1; o >>= 1) mx = fmaxf(mx, __shfl_xor(mx, o, 32));
    float sm = 0.f;
    for (int i = lane; i < SEQ; i += 32) { float e = __expf(row[i] - mx); row[i] = e; sm += e; }
    #pragma unroll
    for (int o = 16; o >= 1; o >>= 1) sm += __shfl_xor(sm, o, 32);
    float inv = 1.f / sm;
    for (int i = lane; i < SEQ; i += 32) sP[r * 264 + i] = (_Float16)(row[i] * inv);
  }

  // ---- O = P @ V (V staged transposed: B-operand K contiguous) ----
  v8f oacc = {};
  for (int kc = 0; kc < 4; ++kc) {
    __syncthreads();
    #pragma unroll
    for (int i = 0; i < 2; ++i) {
      int idx = t + i * 256;
      int r = idx >> 3, c8 = (idx & 7) * 8;            // r=key, c8=dim
      v8h vv = *(const v8h*)(qkv + (rowbase + kc * 64 + r) * 1536 + 2 * DM + h * 64 + c8);
      #pragma unroll
      for (int dd = 0; dd < 8; ++dd) sKV[(c8 + dd) * 72 + r] = vv[dd];
    }
    __syncthreads();
    #pragma unroll
    for (int sub = 0; sub < 2; ++sub) {
      const _Float16* ap = &sP[(rg * 16 + l15) * 264 + kc * 64 + sub * 32 + grp * 8];
      v16h af = cat8(*(const v8h*)ap, *(const v8h*)(ap + 16));
      const _Float16* bp = &sKV[(wc * 16 + l15) * 72 + sub * 32 + grp * 8];
      v16h bf = cat8(*(const v8h*)bp, *(const v8h*)(bp + 16));
      oacc = wmma_f16(af, bf, oacc);
    }
  }
  #pragma unroll
  for (int j = 0; j < 8; ++j) {
    int m = rg * 16 + j + 8 * grp;
    int n = wc * 16 + l15;
    out[(rowbase + q0 + m) * DM + h * 64 + n] = (_Float16)oacc[j];
  }
}

// ---------------------------------------------------------------------------
// Host orchestration (graph-capture safe: kernel launches on `stream` only)
// ---------------------------------------------------------------------------
extern "C" void kernel_launch(void* const* d_in, const int* in_sizes, int n_in,
                              void* d_out, int out_size, void* d_ws, size_t ws_size,
                              hipStream_t stream) {
  (void)in_sizes; (void)n_in; (void)out_size; (void)ws_size;
  const float* patches        = (const float*)d_in[0];
  const float* masked_token   = (const float*)d_in[3];
  const float* partial_params = (const float*)d_in[4];
  const float* emb_w  = (const float*)d_in[5];
  const float* emb_b  = (const float*)d_in[6];
  const float* ln1_g  = (const float*)d_in[7];
  const float* ln1_b  = (const float*)d_in[8];
  const float* in_w   = (const float*)d_in[9];
  const float* in_b   = (const float*)d_in[10];
  const float* op_w   = (const float*)d_in[11];
  const float* op_b   = (const float*)d_in[12];
  const float* rel    = (const float*)d_in[13];
  const float* ln2_g  = (const float*)d_in[14];
  const float* ln2_b  = (const float*)d_in[15];
  const float* fw1    = (const float*)d_in[16];
  const float* fb1    = (const float*)d_in[17];
  const float* fw2    = (const float*)d_in[18];
  const float* fb2    = (const float*)d_in[19];
  const float* outp_w = (const float*)d_in[20];
  const float* outp_b = (const float*)d_in[21];
  float* logits = (float*)d_out;

  // workspace layout (~91 MB)
  char* base = (char*)d_ws;
  float*    proc  = (float*)base;                      base += (size_t)NTOK * 16 * 4;
  float*    x     = (float*)base;                      base += (size_t)NTOK * DM * 4;
  _Float16* xnh   = (_Float16*)base;                   base += (size_t)NTOK * DM * 2;
  _Float16* qkvh  = (_Float16*)base;                   base += (size_t)NTOK * 3 * DM * 2;
  _Float16* attnh = (_Float16*)base;                   base += (size_t)NTOK * DM * 2;
  _Float16* hbh   = (_Float16*)base;                   base += (size_t)NTOK * 2048 * 2;
  _Float16* xh    = (_Float16*)base;                   base += (size_t)NTOK * DM * 2;
  _Float16* wh    = (_Float16*)base;                   // rotating f16 weights, 67MB max

  dim3 blk(256);
  auto cvt = [&](const float* s, _Float16* d, int n) {
    cvt_k<<<dim3((n / 4 + 255) / 256), blk, 0, stream>>>(s, d, n);
  };

  tokenize_k<<<dim3(NTOK / 256), blk, 0, stream>>>(patches, masked_token, partial_params, proc);
  embed_k<<<dim3(NTOK * DM / 256), blk, 0, stream>>>(proc, emb_w, emb_b, x);

  for (int i = 0; i < 6; ++i) {
    layernorm_k<<<dim3(NTOK / 8), blk, 0, stream>>>(x, ln1_g + i * DM, ln1_b + i * DM, xnh);
    cvt(in_w + (size_t)i * 3 * DM * DM, wh, 3 * DM * DM);
    gemm_k<<<dim3(NTOK / 64, 3 * DM / 128), blk, 0, stream>>>(
        xnh, wh, in_b + (size_t)i * 3 * DM, nullptr, nullptr, qkvh,
        NTOK, 3 * DM, DM, 0, 0);
    attention_k<<<dim3(64, SEQ / 32), blk, 0, stream>>>(qkvh, rel + (size_t)i * SEQ * SEQ, attnh);
    cvt(op_w + (size_t)i * DM * DM, wh, DM * DM);
    gemm_k<<<dim3(NTOK / 64, DM / 128), blk, 0, stream>>>(
        attnh, wh, op_b + (size_t)i * DM, x, x, nullptr,
        NTOK, DM, DM, 0, 0);
    layernorm_k<<<dim3(NTOK / 8), blk, 0, stream>>>(x, ln2_g + i * DM, ln2_b + i * DM, xnh);
    cvt(fw1 + (size_t)i * 2048 * DM, wh, 2048 * DM);
    gemm_k<<<dim3(NTOK / 64, 2048 / 128), blk, 0, stream>>>(
        xnh, wh, fb1 + (size_t)i * 2048, nullptr, nullptr, hbh,
        NTOK, 2048, DM, 1 /*GELU*/, 0);
    cvt(fw2 + (size_t)i * DM * 2048, wh, DM * 2048);
    gemm_k<<<dim3(NTOK / 64, DM / 128), blk, 0, stream>>>(
        hbh, wh, fb2 + (size_t)i * DM, x, x, nullptr,
        NTOK, DM, 2048, 0, 0);
  }

  cvt(x, xh, NTOK * DM);
  cvt(outp_w, wh, 65536 * DM);
  // logits: 2048x65536x512, HBM-store-bound; NT stores keep outp_w hot in L2.
  gemm_k<<<dim3(NTOK / 64, 65536 / 128), blk, 0, stream>>>(
      xh, wh, outp_b, nullptr, logits, nullptr,
      NTOK, 65536, DM, 0, 1 /*NT*/);
}
